// MultiheadSelfAttention_81612968559115
// MI455X (gfx1250) — compile-verified, matrix operand order checked by reading, not run on hardware
//
#include <hip/hip_runtime.h>

typedef __attribute__((ext_vector_type(16))) _Float16 v16h;
typedef __attribute__((ext_vector_type(8)))  _Float16 v8h;
typedef __attribute__((ext_vector_type(4)))  _Float16 v4h;
typedef __attribute__((ext_vector_type(8)))  float    v8f;
typedef __attribute__((ext_vector_type(4)))  float    v4f;

#define D_MODEL  1024
#define NHEAD    16
#define HEAD_DIM 64
#define BATCH    4
#define SEQ      2048
#define NROWS    (BATCH * SEQ)   // 8192

__device__ __forceinline__ v16h cat8(v8h lo, v8h hi) {
    v16h r;
#pragma unroll
    for (int i = 0; i < 8; i++) { r[i] = lo[i]; r[8 + i] = hi[i]; }
    return r;
}

__device__ __forceinline__ v8f zero8() {
    v8f r;
#pragma unroll
    for (int i = 0; i < 8; i++) r[i] = 0.0f;
    return r;
}

// ---------------------------------------------------------------------------
// Kernel 1: C[8192 x 1024] = X @ W + b, emitted as f16.
//   VT=false -> dst layout [B][H][S][Dh]   (Q, K)
//   VT=true  -> dst layout [B][H][Dh][S]   (V, pre-transposed for P@V B-frags)
// Block: 256 threads (8 waves), tile M=128 N=128, K-step 32.
// Wave (wm, wn) computes a 32x64 subtile = 8 WMMAs per K-step.
// ---------------------------------------------------------------------------
template <bool VT>
__global__ __launch_bounds__(256)
void qkv_proj_kernel(const float* __restrict__ x,
                     const float* __restrict__ W,
                     const float* __restrict__ bias,
                     _Float16* __restrict__ dst)
{
    const int nBase = blockIdx.x * 128;
    const int mBase = blockIdx.y * 128;
    const int t      = threadIdx.x;
    const int wid    = t >> 5;
    const int lane   = t & 31;
    const int laneLo = lane & 15;
    const int laneHi = lane >> 4;

    __shared__ __align__(16) _Float16 Xs[128][40];  // [m][k] f16, pad 8
    __shared__ __align__(16) _Float16 Ws[128][40];  // [n][k] (transposed) f16

    const int wm = wid >> 1;   // 0..3 : 32 rows each
    const int wn = wid & 1;    // 0..1 : 64 cols each

    v8f acc[2][4];
#pragma unroll
    for (int i = 0; i < 2; i++)
#pragma unroll
        for (int j = 0; j < 4; j++) acc[i][j] = zero8();

    for (int k0 = 0; k0 < D_MODEL; k0 += 32) {
        __syncthreads();
        // Stage X tile 128x32 (f32 -> f16)
#pragma unroll
        for (int i = 0; i < 4; i++) {
            int c   = t + 256 * i;        // 0..1023 (4-float chunks)
            int row = c >> 3;             // 0..127
            int kc  = (c & 7) * 4;        // 0..28
            v4f xv = *(const v4f*)(x + (size_t)(mBase + row) * D_MODEL + k0 + kc);
            v4h xh;
#pragma unroll
            for (int e = 0; e < 4; e++) xh[e] = (_Float16)xv[e];
            *(v4h*)&Xs[row][kc] = xh;
        }
        // Stage W tile 32x128 transposed -> Ws[n][k] (f32 -> f16)
#pragma unroll
        for (int i = 0; i < 4; i++) {
            int c  = t + 256 * i;         // 0..1023
            int kr = c >> 5;              // 0..31
            int nc = (c & 31) * 4;        // 0..124
            v4f wv = *(const v4f*)(W + (size_t)(k0 + kr) * D_MODEL + nBase + nc);
#pragma unroll
            for (int e = 0; e < 4; e++) Ws[nc + e][kr] = (_Float16)wv[e];
        }
        __syncthreads();

        // A-frag layout (16x32 f16): lanes 0-15 K{0..7,16..23}; lanes 16-31 K{8..15,24..31}
        v16h Af[2], Bf[4];
#pragma unroll
        for (int mm = 0; mm < 2; mm++) {
            int row = wm * 32 + mm * 16 + laneLo;
            v8h lo = *(const v8h*)&Xs[row][laneHi * 8];
            v8h hi = *(const v8h*)&Xs[row][16 + laneHi * 8];
            Af[mm] = cat8(lo, hi);
        }
#pragma unroll
        for (int nn = 0; nn < 4; nn++) {
            int col = wn * 64 + nn * 16 + laneLo;
            v8h lo = *(const v8h*)&Ws[col][laneHi * 8];
            v8h hi = *(const v8h*)&Ws[col][16 + laneHi * 8];
            Bf[nn] = cat8(lo, hi);
        }
        __builtin_amdgcn_sched_barrier(0);   // keep all ds_loads issued first
#pragma unroll
        for (int mm = 0; mm < 2; mm++)
#pragma unroll
            for (int nn = 0; nn < 4; nn++)
                acc[mm][nn] = __builtin_amdgcn_wmma_f32_16x16x32_f16(
                    false, Af[mm], false, Bf[nn], (short)0, acc[mm][nn], false, false);
    }

    // Epilogue: add bias, convert f16, scatter to chosen layout
#pragma unroll
    for (int mm = 0; mm < 2; mm++) {
#pragma unroll
        for (int nn = 0; nn < 4; nn++) {
            int col = nBase + wn * 64 + nn * 16 + laneLo;
            float bv = bias[col];
            int h = col >> 6;          // head
            int d = col & 63;          // dim in head
#pragma unroll
            for (int r = 0; r < 8; r++) {
                int row = mBase + wm * 32 + mm * 16 + r + laneHi * 8;
                int bb = row >> 11;    // batch
                int s  = row & 2047;   // seq pos
                float val = acc[mm][nn][r] + bv;
                size_t idx;
                if (VT)
                    idx = (((size_t)bb * NHEAD + h) * HEAD_DIM + d) * SEQ + s;
                else
                    idx = (((size_t)bb * NHEAD + h) * SEQ + s) * HEAD_DIM + d;
                dst[idx] = (_Float16)val;
            }
        }
    }
}

// ---------------------------------------------------------------------------
// Kernel 2: flash attention. Grid (S/64, B*H), 128 threads = 4 waves.
// Each wave owns 16 q rows. Q/K fragments load straight from global; V was
// stored pre-transposed [d][s] so P@V B-fragments are direct b128 global
// loads (L2-resident: 256KB/head). P round-trips through LDS for the
// accumulator->A-fragment layout change. Row sums of P are computed with
// 2 extra WMMAs against an all-ones B tile. sched_barrier(0) pins the
// fragment-load clauses ahead of the WMMA chains for latency overlap.
// 18 WMMAs per wave per 64-key block.
// ---------------------------------------------------------------------------
__global__ __launch_bounds__(128)
void attn_kernel(const _Float16* __restrict__ q,    // [B][H][S][Dh]
                 const _Float16* __restrict__ k,    // [B][H][S][Dh]
                 const _Float16* __restrict__ vt,   // [B][H][Dh][S]
                 float* __restrict__ out)           // [B][S][D_MODEL]
{
    const int qt = blockIdx.x;          // q tile (64 rows)
    const int bh = blockIdx.y;          // b*NHEAD + h
    const int b  = bh >> 4;
    const int h  = bh & 15;
    const int t      = threadIdx.x;
    const int wid    = t >> 5;          // 0..3
    const int lane   = t & 31;
    const int laneLo = lane & 15;
    const int laneHi = lane >> 4;

    const _Float16* qh  = q  + (size_t)bh * SEQ * HEAD_DIM;
    const _Float16* kh  = k  + (size_t)bh * SEQ * HEAD_DIM;
    const _Float16* vth = vt + (size_t)bh * HEAD_DIM * SEQ;

    __shared__ __align__(16) _Float16 plds[4][16][72];   // per-wave P tiles

    // Q fragments: 16 rows x 64 d -> two 16x32 A-frags, straight from global
    const int qrow = qt * 64 + wid * 16 + laneLo;
    v16h Aq[2];
#pragma unroll
    for (int c = 0; c < 2; c++) {
        const _Float16* qp = qh + (size_t)qrow * HEAD_DIM + c * 32;
        v8h lo = *(const v8h*)(qp + laneHi * 8);
        v8h hi = *(const v8h*)(qp + 16 + laneHi * 8);
        Aq[c] = cat8(lo, hi);
    }

    // All-ones B tile: P @ ones gives row sums of P in every accumulator lane
    v16h onesB;
#pragma unroll
    for (int i = 0; i < 16; i++) onesB[i] = (_Float16)1.0f;

    v8f Oacc[4];
#pragma unroll
    for (int n = 0; n < 4; n++) Oacc[n] = zero8();
    float mrow[8], lrow[8];
#pragma unroll
    for (int r = 0; r < 8; r++) { mrow[r] = -3.0e38f; lrow[r] = 0.0f; }
    const float scale = 0.03125f;   // 1/sqrt(D_MODEL)

    for (int j = 0; j < SEQ / 64; ++j) {
        const _Float16* kblk = kh + (size_t)j * 64 * HEAD_DIM;

        if (j + 1 < SEQ / 64) {
            __builtin_prefetch(kblk + 64 * HEAD_DIM, 0, 1);   // global_prefetch_b8
            __builtin_prefetch(vth + (j + 1) * 64, 0, 1);
        }

        // ---- Load ALL K B-fragments up front (one clause, overlapped) ----
        v16h Bk[2][4];
#pragma unroll
        for (int n = 0; n < 4; n++)
#pragma unroll
            for (int c = 0; c < 2; c++) {
                const _Float16* kp = kblk + (size_t)(n * 16 + laneLo) * HEAD_DIM + c * 32;
                v8h lo = *(const v8h*)(kp + laneHi * 8);
                v8h hi = *(const v8h*)(kp + 16 + laneHi * 8);
                Bk[c][n] = cat8(lo, hi);
            }
        __builtin_amdgcn_sched_barrier(0);   // pin loads ahead of WMMAs

        // ---- S = Q @ K^T : 8 back-to-back WMMAs ----
        v8f S[4];
#pragma unroll
        for (int n = 0; n < 4; n++) {
            v8f accn = __builtin_amdgcn_wmma_f32_16x16x32_f16(
                false, Aq[0], false, Bk[0][n], (short)0, zero8(), false, false);
            accn = __builtin_amdgcn_wmma_f32_16x16x32_f16(
                false, Aq[1], false, Bk[1][n], (short)0, accn, false, false);
            S[n] = accn;
        }
#pragma unroll
        for (int n = 0; n < 4; n++)
#pragma unroll
            for (int r = 0; r < 8; r++) S[n][r] *= scale;

        // ---- Online softmax: batched butterfly max-reduce (4 waits, not 32)
        float mx[8];
#pragma unroll
        for (int r = 0; r < 8; r++)
            mx[r] = fmaxf(fmaxf(S[0][r], S[1][r]), fmaxf(S[2][r], S[3][r]));
#pragma unroll
        for (int m = 1; m < 16; m <<= 1) {
            float other[8];
#pragma unroll
            for (int r = 0; r < 8; r++) other[r] = __shfl_xor(mx[r], m, 32);
#pragma unroll
            for (int r = 0; r < 8; r++) mx[r] = fmaxf(mx[r], other[r]);
        }
#pragma unroll
        for (int r = 0; r < 8; r++) {
            float mnew  = fmaxf(mrow[r], mx[r]);
            float alpha = __expf(mrow[r] - mnew);
            mrow[r] = mnew;
            lrow[r] *= alpha;
#pragma unroll
            for (int n = 0; n < 4; n++) {
                S[n][r] = __expf(S[n][r] - mnew);
                Oacc[n][r] *= alpha;
            }
        }

        // ---- P (C layout) -> LDS f16 -> A-fragment layout ----
#pragma unroll
        for (int n = 0; n < 4; n++)
#pragma unroll
            for (int r = 0; r < 8; r++)
                plds[wid][r + laneHi * 8][n * 16 + laneLo] = (_Float16)S[n][r];
        __syncthreads();        // flush DS stores (cross-lane within wave)

        // ---- Load A frags + ALL V B-fragments up front ----
        v16h Ap[2];
#pragma unroll
        for (int c = 0; c < 2; c++) {
            v8h lo = *(const v8h*)&plds[wid][laneLo][c * 32 + laneHi * 8];
            v8h hi = *(const v8h*)&plds[wid][laneLo][c * 32 + 16 + laneHi * 8];
            Ap[c] = cat8(lo, hi);
        }
        v16h Bv[2][4];
#pragma unroll
        for (int c = 0; c < 2; c++)
#pragma unroll
            for (int n = 0; n < 4; n++) {
                const _Float16* vp = vth + (size_t)(n * 16 + laneLo) * SEQ
                                         + j * 64 + c * 32;
                v8h blo = *(const v8h*)(vp + laneHi * 8);
                v8h bhi = *(const v8h*)(vp + 16 + laneHi * 8);
                Bv[c][n] = cat8(blo, bhi);
            }
        __builtin_amdgcn_sched_barrier(0);   // pin loads ahead of WMMAs

        // ---- O += P @ V  and  lrow += P @ ones (row sums) : 10 WMMAs ----
        v8f lacc = zero8();
#pragma unroll
        for (int c = 0; c < 2; c++) {
            lacc = __builtin_amdgcn_wmma_f32_16x16x32_f16(
                false, Ap[c], false, onesB, (short)0, lacc, false, false);
#pragma unroll
            for (int n = 0; n < 4; n++)
                Oacc[n] = __builtin_amdgcn_wmma_f32_16x16x32_f16(
                    false, Ap[c], false, Bv[c][n], (short)0, Oacc[n], false, false);
        }
#pragma unroll
        for (int r = 0; r < 8; r++) lrow[r] += lacc[r];

        __syncthreads();        // plds reusable next iteration
    }

    // Normalize and write out [B][S][D_MODEL] f32
#pragma unroll
    for (int n = 0; n < 4; n++) {
#pragma unroll
        for (int r = 0; r < 8; r++) {
            int s   = qt * 64 + wid * 16 + r + laneHi * 8;
            int col = h * HEAD_DIM + n * 16 + laneLo;
            out[((size_t)b * SEQ + s) * D_MODEL + col] = Oacc[n][r] / lrow[r];
        }
    }
}

extern "C" void kernel_launch(void* const* d_in, const int* in_sizes, int n_in,
                              void* d_out, int out_size, void* d_ws, size_t ws_size,
                              hipStream_t stream) {
    const float* x  = (const float*)d_in[0];
    const float* Wq = (const float*)d_in[1];
    const float* bq = (const float*)d_in[2];
    const float* Wk = (const float*)d_in[3];
    const float* bk = (const float*)d_in[4];
    const float* Wv = (const float*)d_in[5];
    const float* bv = (const float*)d_in[6];
    float* out = (float*)d_out;

    const size_t per = (size_t)BATCH * NHEAD * SEQ * HEAD_DIM;  // 8.4M elems
    _Float16* qws = (_Float16*)d_ws;
    _Float16* kws = qws + per;
    _Float16* vws = kws + per;     // total 48MB of f16 in d_ws

    dim3 g1(D_MODEL / 128, NROWS / 128);   // (8, 64)
    qkv_proj_kernel<false><<<g1, 256, 0, stream>>>(x, Wq, bq, qws);
    qkv_proj_kernel<false><<<g1, 256, 0, stream>>>(x, Wk, bk, kws);
    qkv_proj_kernel<true ><<<g1, 256, 0, stream>>>(x, Wv, bv, vws);

    dim3 g2(SEQ / 64, BATCH * NHEAD);      // (32, 64)
    attn_kernel<<<g2, 128, 0, stream>>>(qws, kws, vws, out);
}
